// Net_66649302499630
// MI455X (gfx1250) — compile-verified, matrix operand order checked by reading, not run on hardware
//
#include <hip/hip_runtime.h>
#include <hip/hip_bf16.h>
#include <stdint.h>

// ---------------------------------------------------------------------------
// Types for CDNA5 WMMA (wave32). Activations & weights stored as bf16 (ushort),
// all arithmetic in fp32, matrix math via v_wmma_f32_16x16x32_bf16.
// ---------------------------------------------------------------------------
typedef __attribute__((ext_vector_type(16))) __bf16 v16bf;
typedef __attribute__((ext_vector_type(8)))  float  v8f;
typedef unsigned short bfu;

#define BM 128
#define BN 128
#define BKK 32
#define GEMM_THREADS 256

__device__ __forceinline__ bfu f2bf(float f) {
  union { float f; uint32_t u; } c; c.f = f;
  uint32_t u = c.u + 0x7FFFu + ((c.u >> 16) & 1u);   // round-to-nearest-even
  return (bfu)(u >> 16);
}
__device__ __forceinline__ float bf2f(bfu h) {
  union { uint32_t u; float f; } c; c.u = (uint32_t)h << 16; return c.f;
}

// Fragment per CDNA5 ISA (16-bit A 16x32 / B 32x16): lanes 0-15 hold
// K = {hi*8+0..7, 16+hi*8+0..7}, hi = lane>>4. Row (A) / col (B, N-major LDS)
// runs are contiguous -> fragment = two 16-byte LDS reads.
__device__ __forceinline__ v16bf frag_from_lds(const bfu* rowp, int hi) {
  union { uint32_t u[8]; v16bf v; } r;
  const uint32_t* p0 = (const uint32_t*)(rowp + hi * 8);
  const uint32_t* p1 = (const uint32_t*)(rowp + 16 + hi * 8);
#pragma unroll
  for (int i = 0; i < 4; ++i) { r.u[i] = p0[i]; r.u[4 + i] = p1[i]; }
  return r.v;
}

// ---------------------------------------------------------------------------
// Fused GEMM: Y[M,N] = act(X[M,K] @ W[K,N] + bias); X row stride ldx (mult of 8,
// zero-padded cols beyond K), W is [K,N] bf16, Y bf16 stride N.
// LDS double-buffered; interior A-tile staged with global_load_async_to_lds_b128
// (ASYNCcnt), B-tile transposed through VGPRs, fragments via ds_load_b128.
// ---------------------------------------------------------------------------
__global__ __launch_bounds__(GEMM_THREADS)
void gemm_bf16_wmma(const bfu* __restrict__ X, int ldx,
                    const bfu* __restrict__ W, const float* __restrict__ bias,
                    bfu* __restrict__ Y, int M, int Kd, int Nc, int relu) {
  __shared__ bfu AsBuf[2 * BM * BKK];   // [buf][row][k]
  __shared__ bfu BsBuf[2 * BN * BKK];   // [buf][col][k]  (N-major)
  const int m0 = blockIdx.y * BM;
  const int n0 = blockIdx.x * BN;
  const int lane = threadIdx.x & 31;
  const int wave = threadIdx.x >> 5;
  const int wm = wave & 3;            // 4 waves along M -> 32 rows each
  const int wn = wave >> 2;           // 2 waves along N -> 64 cols each
  const int lh = lane & 15;
  const int hi = lane >> 4;
  const bool mnFull = (m0 + BM <= M) && (n0 + BN <= Nc);
  const int ktiles = (Kd + BKK - 1) / BKK;

  v8f acc[2][4] = {};

  auto stage = [&](int kt, int buf) {
    const int k0 = kt * BKK;
    bfu* As = AsBuf + buf * (BM * BKK);
    bfu* Bs = BsBuf + buf * (BN * BKK);
    if (mnFull && (k0 + BKK <= Kd)) {
      // ---- fast interior path ----
#pragma unroll
      for (int i = 0; i < 2; ++i) {
        int idx = (int)threadIdx.x + GEMM_THREADS * i;   // 0..511
        int row = idx >> 2;                              // 0..127
        int col = (idx & 3) * 8;                         // 0,8,16,24
        uint64_t gaddr = (uint64_t)(uintptr_t)&X[(size_t)(m0 + row) * ldx + k0 + col];
        uint32_t laddr = (uint32_t)(uintptr_t)&As[row * BKK + col];
        // async 16B global->LDS copy, tracked by ASYNCcnt
        asm volatile("global_load_async_to_lds_b128 %0, %1, off"
                     :: "v"(laddr), "v"(gaddr) : "memory");
      }
#pragma unroll
      for (int i = 0; i < 2; ++i) {
        int idx = (int)threadIdx.x + GEMM_THREADS * i;
        int k   = idx >> 4;                              // 0..31
        int col = (idx & 15) * 8;                        // 0..120
        uint4 w4 = *(const uint4*)&W[(size_t)(k0 + k) * Nc + n0 + col];
        const bfu* ws = (const bfu*)&w4;
#pragma unroll
        for (int j = 0; j < 8; ++j) Bs[(col + j) * BKK + k] = ws[j];
      }
      if (k0 + 2 * BKK <= Kd)                            // global_prefetch_b8 hint (W stream)
        __builtin_prefetch(&W[(size_t)(k0 + BKK + ((int)threadIdx.x >> 3)) * Nc + n0], 0, 0);
    } else {
      // ---- edge path: per-element predicated (compact, no unroll) ----
#pragma unroll 1
      for (int i = 0; i < (BM * BKK) / GEMM_THREADS; ++i) {
        int e = (int)threadIdx.x + GEMM_THREADS * i;
        int row = e >> 5, col = e & 31;
        int gm = m0 + row, gk = k0 + col;
        As[row * BKK + col] = (gm < M && gk < Kd) ? X[(size_t)gm * ldx + gk] : (bfu)0;
      }
#pragma unroll 1
      for (int i = 0; i < (BN * BKK) / GEMM_THREADS; ++i) {
        int e = (int)threadIdx.x + GEMM_THREADS * i;
        int k = e >> 7, col = e & 127;
        int gn = n0 + col, gk = k0 + k;
        Bs[col * BKK + k] = (gn < Nc && gk < Kd) ? W[(size_t)gk * Nc + gn] : (bfu)0;
      }
    }
  };

  stage(0, 0);
  for (int kt = 0; kt < ktiles; ++kt) {
    const int buf = kt & 1;
    asm volatile("s_wait_asynccnt 0x0" ::: "memory");    // own async copies done
    __syncthreads();                                     // all waves' staging visible
    const bfu* As = AsBuf + buf * (BM * BKK);
    const bfu* Bs = BsBuf + buf * (BN * BKK);

    v16bf afrag[2], bfrag[4];
#pragma unroll
    for (int mt = 0; mt < 2; ++mt)
      afrag[mt] = frag_from_lds(&As[(wm * 32 + mt * 16 + lh) * BKK], hi);
#pragma unroll
    for (int nt = 0; nt < 4; ++nt)
      bfrag[nt] = frag_from_lds(&Bs[(wn * 64 + nt * 16 + lh) * BKK], hi);

    if (kt + 1 < ktiles) stage(kt + 1, buf ^ 1);         // overlap next tile fill

#pragma unroll
    for (int mt = 0; mt < 2; ++mt)
#pragma unroll
      for (int nt = 0; nt < 4; ++nt)
        acc[mt][nt] = __builtin_amdgcn_wmma_f32_16x16x32_bf16(
            false, afrag[mt], false, bfrag[nt], (short)0, acc[mt][nt], false, false);
  }

  // Epilogue: C/D layout — VGPR r holds M=r (lanes 0-15) / M=8+r (lanes 16-31)
#pragma unroll
  for (int mt = 0; mt < 2; ++mt) {
#pragma unroll
    for (int nt = 0; nt < 4; ++nt) {
      int col = n0 + wn * 64 + nt * 16 + lh;
      float bv = (bias != nullptr && col < Nc) ? bias[col] : 0.f;
#pragma unroll
      for (int r = 0; r < 8; ++r) {
        int row = m0 + wm * 32 + mt * 16 + hi * 8 + r;
        if (row < M && col < Nc) {
          float v = acc[mt][nt][r] + bv;
          if (relu) v = fmaxf(v, 0.f);
          Y[(size_t)row * Nc + col] = f2bf(v);
        }
      }
    }
  }
}

// ---------------------------------------------------------------------------
// fp32 -> bf16 conversion (weights, pos)
// ---------------------------------------------------------------------------
__global__ void cvt_f32_bf16(const float* __restrict__ src, size_t n, bfu* __restrict__ dst) {
  for (size_t i = (size_t)blockIdx.x * blockDim.x + threadIdx.x; i < n;
       i += (size_t)gridDim.x * blockDim.x)
    dst[i] = f2bf(src[i]);
}

// ---------------------------------------------------------------------------
// BatchNorm: column stats (two-pass with atomics) + affine (bf16 storage)
// ---------------------------------------------------------------------------
__global__ void colstats_partial(const bfu* __restrict__ Y, int M, int Nc, int rowsPer,
                                 float* __restrict__ sum, float* __restrict__ sumsq) {
  int col = blockIdx.x * blockDim.x + threadIdx.x;
  if (col >= Nc) return;
  int r0 = blockIdx.y * rowsPer;
  int r1 = r0 + rowsPer; if (r1 > M) r1 = M;
  float s = 0.f, q = 0.f;
  for (int r = r0; r < r1; ++r) { float v = bf2f(Y[(size_t)r * Nc + col]); s += v; q += v * v; }
  atomicAdd(&sum[col], s);
  atomicAdd(&sumsq[col], q);
}

__global__ void colstats_final(const float* __restrict__ sum, const float* __restrict__ sumsq,
                               int M, int Nc, float* __restrict__ mean, float* __restrict__ rstd) {
  int col = blockIdx.x * blockDim.x + threadIdx.x;
  if (col >= Nc) return;
  float m = sum[col] / (float)M;
  float v = sumsq[col] / (float)M - m * m;
  if (v < 0.f) v = 0.f;
  mean[col] = m;
  rstd[col] = rsqrtf(v + 1e-5f);
}

__global__ void bn_affine(bfu* __restrict__ Y, size_t total, int Nc,
                          const float* __restrict__ mean, const float* __restrict__ rstd,
                          const float* __restrict__ g, const float* __restrict__ be, int relu) {
  for (size_t i = (size_t)blockIdx.x * blockDim.x + threadIdx.x; i < total;
       i += (size_t)gridDim.x * blockDim.x) {
    int c = (int)(i % (size_t)Nc);
    float v = (bf2f(Y[i]) - mean[c]) * rstd[c] * g[c] + be[c];
    if (relu) v = fmaxf(v, 0.f);
    Y[i] = f2bf(v);
  }
}

// ---------------------------------------------------------------------------
// Elementwise helpers (bf16 storage)
// ---------------------------------------------------------------------------
__global__ void dwscale(const bfu* __restrict__ X, const float* __restrict__ w,
                        const float* __restrict__ b, size_t total, int Nc, bfu* __restrict__ Y) {
  for (size_t i = (size_t)blockIdx.x * blockDim.x + threadIdx.x; i < total;
       i += (size_t)gridDim.x * blockDim.x) {
    int c = (int)(i % (size_t)Nc);
    Y[i] = f2bf(bf2f(X[i]) * w[c] + b[c]);
  }
}

__global__ void add_relu(const bfu* __restrict__ A, const bfu* __restrict__ B,
                         size_t total, bfu* __restrict__ Y) {
  for (size_t i = (size_t)blockIdx.x * blockDim.x + threadIdx.x; i < total;
       i += (size_t)gridDim.x * blockDim.x)
    Y[i] = f2bf(fmaxf(bf2f(A[i]) + bf2f(B[i]), 0.f));
}

__global__ void copy2d(const bfu* __restrict__ src, int rows, int cols,
                       bfu* __restrict__ dst, int ldd, int c0) {
  size_t total = (size_t)rows * cols;
  for (size_t i = (size_t)blockIdx.x * blockDim.x + threadIdx.x; i < total;
       i += (size_t)gridDim.x * blockDim.x) {
    int r = (int)(i / (size_t)cols), c = (int)(i % (size_t)cols);
    dst[(size_t)r * ldd + c0 + c] = src[i];
  }
}

__global__ void copy2d_f32(const float* __restrict__ src, int rows, int cols,
                           bfu* __restrict__ dst, int ldd, int c0) {
  size_t total = (size_t)rows * cols;
  for (size_t i = (size_t)blockIdx.x * blockDim.x + threadIdx.x; i < total;
       i += (size_t)gridDim.x * blockDim.x) {
    int r = (int)(i / (size_t)cols), c = (int)(i % (size_t)cols);
    dst[(size_t)r * ldd + c0 + c] = f2bf(src[i]);
  }
}

__global__ void zerofill2d(bfu* __restrict__ dst, int rows, int ld, int c0, int c1) {
  int w = c1 - c0;
  size_t total = (size_t)rows * w;
  for (size_t i = (size_t)blockIdx.x * blockDim.x + threadIdx.x; i < total;
       i += (size_t)gridDim.x * blockDim.x) {
    int r = (int)(i / (size_t)w), c = (int)(i % (size_t)w);
    dst[(size_t)r * ld + c0 + c] = 0;
  }
}

__global__ void build_pos4(const float* __restrict__ p, const float* __restrict__ refl,
                           const float* __restrict__ sf, const int* __restrict__ bat,
                           int n, float* __restrict__ out) {
  int i = blockIdx.x * blockDim.x + threadIdx.x;
  if (i >= n) return;
  float s = sf[bat[i]];
  out[i * 4 + 0] = p[i * 3 + 0] / s;
  out[i * 4 + 1] = p[i * 3 + 1] / s;
  out[i * 4 + 2] = p[i * 3 + 2] / s;
  out[i * 4 + 3] = refl[i];
}

__global__ void gather_pbr(const float* __restrict__ p_in, const int* __restrict__ b_in,
                           const float* __restrict__ r_in, const int* __restrict__ idx,
                           int n, float* __restrict__ p_out, int* __restrict__ b_out,
                           float* __restrict__ r_out) {
  int i = blockIdx.x * blockDim.x + threadIdx.x;
  if (i >= n) return;
  int j = idx[i];
  p_out[i * 3 + 0] = p_in[j * 3 + 0];
  p_out[i * 3 + 1] = p_in[j * 3 + 1];
  p_out[i * 3 + 2] = p_in[j * 3 + 2];
  b_out[i] = b_in[j];
  r_out[i] = r_in[j];
}

// msg[row] = concat(x[src], pos4[src]-pos4[center]); pads cols [xc+4, ldm) with 0
__global__ void build_msg(const bfu* __restrict__ x, int xc, const float* __restrict__ pos4,
                          const int* __restrict__ esrc, const int* __restrict__ idx,
                          int Kn, int ldm, bfu* __restrict__ msg) {
  int row = blockIdx.x;
  int c = row / Kn;
  int s = esrc[row];
  int ctr = idx[c];
  int w = xc + 4;
  for (int t = threadIdx.x; t < ldm; t += blockDim.x) {
    bfu v;
    if (t < xc)      v = x[(size_t)s * xc + t];
    else if (t < w)  v = f2bf(pos4[s * 4 + (t - xc)] - pos4[ctr * 4 + (t - xc)]);
    else             v = 0;
    msg[(size_t)row * ldm + t] = v;
  }
}

__global__ void kmax_kernel(const bfu* __restrict__ h, int Nc, int Kn, int ch,
                            bfu* __restrict__ out) {
  size_t total = (size_t)Nc * ch;
  for (size_t i = (size_t)blockIdx.x * blockDim.x + threadIdx.x; i < total;
       i += (size_t)gridDim.x * blockDim.x) {
    int c = (int)(i / (size_t)ch), d = (int)(i % (size_t)ch);
    float m = -INFINITY;
    for (int j = 0; j < Kn; ++j)
      m = fmaxf(m, bf2f(h[((size_t)(c * Kn + j)) * ch + d]));
    out[i] = f2bf(m);
  }
}

__global__ void segmax_kernel(const bfu* __restrict__ h, int Nr, int nseg, int ch,
                              bfu* __restrict__ out) {
  int i = blockIdx.x * blockDim.x + threadIdx.x;
  if (i >= nseg * ch) return;
  int s = i / ch, d = i % ch;
  int per = Nr / nseg;
  float m = -INFINITY;
  for (int r = s * per; r < (s + 1) * per; ++r)
    m = fmaxf(m, bf2f(h[(size_t)r * ch + d]));
  out[i] = f2bf(m);
}

// 2-NN inverse-distance interpolation into columns [c0, c0+D) with stride ldo
__global__ void interp_kernel(const bfu* __restrict__ xs, const float* __restrict__ ps,
                              const float* __restrict__ pt, const int* __restrict__ nidx,
                              int T, int D, bfu* __restrict__ out, int ldo, int c0) {
  size_t total = (size_t)T * D;
  for (size_t i = (size_t)blockIdx.x * blockDim.x + threadIdx.x; i < total;
       i += (size_t)gridDim.x * blockDim.x) {
    int t = (int)(i / (size_t)D), d = (int)(i % (size_t)D);
    int i0 = nidx[t * 2 + 0], i1 = nidx[t * 2 + 1];
    float d0 = 0.f, d1 = 0.f;
    for (int a = 0; a < 3; ++a) {
      float x0 = pt[t * 3 + a] - ps[i0 * 3 + a]; d0 += x0 * x0;
      float x1 = pt[t * 3 + a] - ps[i1 * 3 + a]; d1 += x1 * x1;
    }
    float w0 = 1.f / fmaxf(d0, 1e-16f), w1 = 1.f / fmaxf(d1, 1e-16f);
    float ws = w0 + w1; w0 /= ws; w1 /= ws;
    out[(size_t)t * ldo + c0 + d] =
        f2bf(w0 * bf2f(xs[(size_t)i0 * D + d]) + w1 * bf2f(xs[(size_t)i1 * D + d]));
  }
}

// head: out[2, M] = (h[M,512] @ W2[512,2] + b2).T  (fp32 out)
__global__ void head2_kernel(const bfu* __restrict__ h, const float* __restrict__ W2,
                             const float* __restrict__ b2, int M, int Kd,
                             float* __restrict__ out) {
  int r = blockIdx.x * blockDim.x + threadIdx.x;
  if (r >= M) return;
  float a0 = b2[0], a1 = b2[1];
  for (int k = 0; k < Kd; ++k) {
    float v = bf2f(h[(size_t)r * Kd + k]);
    a0 += v * W2[k * 2 + 0];
    a1 += v * W2[k * 2 + 1];
  }
  out[r] = a0;
  out[(size_t)M + r] = a1;
}

// ---------------------------------------------------------------------------
// Orchestration
// ---------------------------------------------------------------------------
static inline unsigned gblocks(size_t total) {
  size_t b = (total + 255) / 256;
  if (b > 131072) b = 131072;
  return (unsigned)b;
}

extern "C" void kernel_launch(void* const* d_in, const int* in_sizes, int n_in,
                              void* d_out, int out_size, void* d_ws, size_t ws_size,
                              hipStream_t stream) {
  (void)n_in; (void)out_size; (void)ws_size;
  const int N = 32768, Bn = 4, Kn = 32;
  const int N1 = N / 2, N2 = N / 4, N3 = N / 8;

  auto F = [&](int i) { return (const float*)d_in[i]; };
  auto I = [&](int i) { return (const int*)d_in[i]; };

  // Param leaf indices per JAX sorted-key pytree flattening
  const int FP1 = 0, FP2 = 6, FP3 = 12, FP4 = 18, GSA = 24, HEAD = 30;
  const int SA1L = 36, SA1R = 42, SA2L = 70, SA2R = 76, SA3L = 104, SA3R = 110, STEM = 138;
  const int POS = 140, REFL = 141, SF = 142, BATCH = 143;
  const int IDX1 = 144, IDX2 = 145, IDX3 = 146, E1 = 147, E2 = 148, E3 = 149;
  const int F4N = 150, F3N = 151, F2N = 152, F1N = 153;

  // ---- workspace carve ----
  char* wp = (char*)d_ws;
  auto alloc = [&](size_t bytes) { char* p = wp; wp += (bytes + 255) & ~(size_t)255; return p; };

  // bf16 weight copies (converted once per launch, same stream => ordered)
  bfu* WB[160] = {};
  auto cvtW = [&](int i) {
    size_t n = (size_t)in_sizes[i];
    bfu* p = (bfu*)alloc(n * 2);
    cvt_f32_bf16<<<gblocks(n), 256, 0, stream>>>(F(i), n, p);
    WB[i] = p;
  };
  {
    const int wl[] = {FP1+0,FP1+4, FP2+0,FP2+4, FP3+0,FP3+4, FP4+0,FP4+4,
                      GSA+0,GSA+4, HEAD+2,
                      SA1L+0,SA1L+4, SA2L+0,SA2L+4, SA3L+0,SA3L+4, STEM+0,
                      SA1R+20,SA1R+8,SA1R+16,SA1R+24,
                      SA2R+20,SA2R+8,SA2R+16,SA2R+24,
                      SA3R+20,SA3R+8,SA3R+16,SA3R+24};
    for (int i = 0; i < (int)(sizeof(wl)/sizeof(wl[0])); ++i) cvtW(wl[i]);
  }

  bfu* posbf   = (bfu*)alloc((size_t)N * 3 * 2);
  bfu* x0      = (bfu*)alloc((size_t)N * 32 * 2);
  float* pos4_0 = (float*)alloc((size_t)N * 4 * 4);
  float* p1 = (float*)alloc((size_t)N1 * 3 * 4); int* b1 = (int*)alloc((size_t)N1 * 4);
  float* r1 = (float*)alloc((size_t)N1 * 4);     float* pos4_1 = (float*)alloc((size_t)N1 * 4 * 4);
  bfu* x1 = (bfu*)alloc((size_t)N1 * 128 * 2);
  float* p2 = (float*)alloc((size_t)N2 * 3 * 4); int* b2v = (int*)alloc((size_t)N2 * 4);
  float* r2 = (float*)alloc((size_t)N2 * 4);     float* pos4_2 = (float*)alloc((size_t)N2 * 4 * 4);
  bfu* x2 = (bfu*)alloc((size_t)N2 * 256 * 2);
  float* p3 = (float*)alloc((size_t)N3 * 3 * 4); int* b3v = (int*)alloc((size_t)N3 * 4);
  float* r3 = (float*)alloc((size_t)N3 * 4);
  bfu* x3 = (bfu*)alloc((size_t)N3 * 512 * 2);
  bfu* xg = (bfu*)alloc((size_t)Bn * 512 * 2);
  float* pg = (float*)alloc((size_t)Bn * 3 * 4);
  float* statsum = (float*)alloc(2048 * 4);
  float* statsq  = (float*)alloc(2048 * 4);
  float* meanb   = (float*)alloc(2048 * 4);
  float* rstdb   = (float*)alloc(2048 * 4);
  bfu* arenaA = (bfu*)alloc((size_t)524288 * 136 * 2);
  bfu* arenaB = (bfu*)alloc((size_t)524288 * 136 * 2);

  // ---- building blocks ----
  auto gemm = [&](const bfu* X, int ldx, int biW, int biB, bfu* Y,
                  int M, int Kd, int Nc, bool relu) {
    dim3 g((Nc + BN - 1) / BN, (M + BM - 1) / BM);
    gemm_bf16_wmma<<<g, GEMM_THREADS, 0, stream>>>(X, ldx, WB[biW],
        biB >= 0 ? F(biB) : nullptr, Y, M, Kd, Nc, relu ? 1 : 0);
  };
  auto bnorm = [&](bfu* Y, int M, int Nc, int biG, int biBe, bool relu) {
    hipMemsetAsync(statsum, 0, (size_t)Nc * 4, stream);
    hipMemsetAsync(statsq,  0, (size_t)Nc * 4, stream);
    dim3 g((Nc + 255) / 256, (M + 2047) / 2048);
    colstats_partial<<<g, 256, 0, stream>>>(Y, M, Nc, 2048, statsum, statsq);
    colstats_final<<<(Nc + 255) / 256, 256, 0, stream>>>(statsum, statsq, M, Nc, meanb, rstdb);
    size_t tot = (size_t)M * Nc;
    bn_affine<<<gblocks(tot), 256, 0, stream>>>(Y, tot, Nc, meanb, rstdb, F(biG), F(biBe),
                                                relu ? 1 : 0);
  };
  auto dsblock = [&](bfu* h, int rows, int e, int qi, bfu* tmp) {
    size_t tot = (size_t)rows * e;
    dwscale<<<gblocks(tot), 256, 0, stream>>>(h, F(qi + 3), F(qi + 0), tot, e, tmp);
    bnorm(tmp, rows, e, qi + 2, qi + 1, true);
    gemm(tmp, e, qi + 4, qi + 5, h, rows, e, e, false);
    bnorm(h, rows, e, qi + 7, qi + 6, true);
  };
  auto resblock = [&](bfu* x, int rows, int c, int ri, bfu* A, bfu* B) {
    int e = 4 * c;
    gemm(x, c, ri + 20, ri + 21, A, rows, c, e, false);
    bnorm(A, rows, e, ri + 23, ri + 22, true);
    dsblock(A, rows, e, ri + 4, B);
    bnorm(A, rows, e, ri + 1, ri + 0, true);
    dsblock(A, rows, e, ri + 12, B);
    bnorm(A, rows, e, ri + 3, ri + 2, false);
    gemm(A, e, ri + 24, ri + 25, B, rows, e, c, false);
    bnorm(B, rows, c, ri + 27, ri + 26, false);
    size_t tot = (size_t)rows * c;
    add_relu<<<gblocks(tot), 256, 0, stream>>>(B, x, tot, x);
  };
  auto sastage = [&](const bfu* xin, int cin, const float* pos4in,
                     const int* idx, const int* esrc, int Nc_,
                     int biL, int c1, int c2, int ri, bfu* xout) {
    int rows = Nc_ * Kn;
    int ldm = (cin + 4 + 7) & ~7;   // pad K to multiple of 8 for aligned staging
    build_msg<<<rows, 64, 0, stream>>>(xin, cin, pos4in, esrc, idx, Kn, ldm, arenaA);
    gemm(arenaA, ldm, biL + 0, biL + 1, arenaB, rows, cin + 4, c1, true);
    gemm(arenaB, c1, biL + 4, biL + 5, arenaA, rows, c1, c2, true);
    bnorm(arenaA, rows, c2, biL + 3, biL + 2, false);
    kmax_kernel<<<gblocks((size_t)Nc_ * c2), 256, 0, stream>>>(arenaA, Nc_, Kn, c2, xout);
    resblock(xout, Nc_, c2, ri, arenaA, arenaB);
  };

  // ---- network ----
  cvt_f32_bf16<<<gblocks((size_t)N * 3), 256, 0, stream>>>(F(POS), (size_t)N * 3, posbf);
  gemm(posbf, 3, STEM + 0, STEM + 1, x0, N, 3, 32, true);
  build_pos4<<<(N + 255) / 256, 256, 0, stream>>>(F(POS), F(REFL), F(SF), I(BATCH), N, pos4_0);

  // SA1 -> x1[N1,128]
  sastage(x0, 32, pos4_0, I(IDX1), I(E1), N1, SA1L, 64, 128, SA1R, x1);
  gather_pbr<<<(N1 + 255) / 256, 256, 0, stream>>>(F(POS), I(BATCH), F(REFL), I(IDX1), N1, p1, b1, r1);
  build_pos4<<<(N1 + 255) / 256, 256, 0, stream>>>(p1, r1, F(SF), b1, N1, pos4_1);

  // SA2 -> x2[N2,256]
  sastage(x1, 128, pos4_1, I(IDX2), I(E2), N2, SA2L, 192, 256, SA2R, x2);
  gather_pbr<<<(N2 + 255) / 256, 256, 0, stream>>>(p1, b1, r1, I(IDX2), N2, p2, b2v, r2);
  build_pos4<<<(N2 + 255) / 256, 256, 0, stream>>>(p2, r2, F(SF), b2v, N2, pos4_2);

  // SA3 -> x3[N3,512]
  sastage(x2, 256, pos4_2, I(IDX3), I(E3), N3, SA3L, 384, 512, SA3R, x3);
  gather_pbr<<<(N3 + 255) / 256, 256, 0, stream>>>(p2, b2v, r2, I(IDX3), N3, p3, b3v, r3);

  // gsa: mlp(concat(x3, p3)) then segment max -> xg[B,512]
  copy2d<<<gblocks((size_t)N3 * 512), 256, 0, stream>>>(x3, N3, 512, arenaA, 520, 0);
  copy2d_f32<<<gblocks((size_t)N3 * 3), 256, 0, stream>>>(p3, N3, 3, arenaA, 520, 512);
  zerofill2d<<<gblocks((size_t)N3 * 5), 256, 0, stream>>>(arenaA, N3, 520, 515, 520);
  gemm(arenaA, 520, GSA + 0, GSA + 1, arenaB, N3, 515, 512, true);
  gemm(arenaB, 512, GSA + 4, GSA + 5, arenaA, N3, 512, 512, true);
  bnorm(arenaA, N3, 512, GSA + 3, GSA + 2, false);
  segmax_kernel<<<(Bn * 512 + 255) / 256, 256, 0, stream>>>(arenaA, N3, Bn, 512, xg);

  // fp4
  hipMemsetAsync(pg, 0, (size_t)Bn * 3 * 4, stream);
  interp_kernel<<<gblocks((size_t)N3 * 512), 256, 0, stream>>>(xg, pg, p3, I(F4N), N3, 512,
                                                               arenaB, 1024, 0);
  copy2d<<<gblocks((size_t)N3 * 512), 256, 0, stream>>>(x3, N3, 512, arenaB, 1024, 512);
  gemm(arenaB, 1024, FP4 + 0, FP4 + 1, arenaA, N3, 1024, 768, true);
  gemm(arenaA, 768, FP4 + 4, FP4 + 5, arenaB, N3, 768, 512, true);
  bnorm(arenaB, N3, 512, FP4 + 3, FP4 + 2, false);            // x in arenaB [N3,512]

  // fp3
  interp_kernel<<<gblocks((size_t)N2 * 512), 256, 0, stream>>>(arenaB, p3, p2, I(F3N), N2, 512,
                                                               arenaA, 768, 0);
  copy2d<<<gblocks((size_t)N2 * 256), 256, 0, stream>>>(x2, N2, 256, arenaA, 768, 512);
  gemm(arenaA, 768, FP3 + 0, FP3 + 1, arenaB, N2, 768, 640, true);
  gemm(arenaB, 640, FP3 + 4, FP3 + 5, arenaA, N2, 640, 512, true);
  bnorm(arenaA, N2, 512, FP3 + 3, FP3 + 2, false);            // x in arenaA [N2,512]

  // fp2
  interp_kernel<<<gblocks((size_t)N1 * 512), 256, 0, stream>>>(arenaA, p2, p1, I(F2N), N1, 512,
                                                               arenaB, 640, 0);
  copy2d<<<gblocks((size_t)N1 * 128), 256, 0, stream>>>(x1, N1, 128, arenaB, 640, 512);
  gemm(arenaB, 640, FP2 + 0, FP2 + 1, arenaA, N1, 640, 512, true);
  gemm(arenaA, 512, FP2 + 4, FP2 + 5, arenaB, N1, 512, 512, true);
  bnorm(arenaB, N1, 512, FP2 + 3, FP2 + 2, false);            // x in arenaB [N1,512]

  // fp1
  interp_kernel<<<gblocks((size_t)N * 512), 256, 0, stream>>>(arenaB, p1, F(POS), I(F1N), N, 512,
                                                              arenaA, 544, 0);
  copy2d<<<gblocks((size_t)N * 32), 256, 0, stream>>>(x0, N, 32, arenaA, 544, 512);
  gemm(arenaA, 544, FP1 + 0, FP1 + 1, arenaB, N, 544, 512, true);
  gemm(arenaB, 512, FP1 + 4, FP1 + 5, arenaA, N, 512, 512, true);
  bnorm(arenaA, N, 512, FP1 + 3, FP1 + 2, false);             // x in arenaA [N,512]

  // head
  gemm(arenaA, 512, HEAD + 2, HEAD + 3, arenaB, N, 512, 512, false);
  bnorm(arenaB, N, 512, HEAD + 1, HEAD + 0, true);
  head2_kernel<<<(N + 255) / 256, 256, 0, stream>>>(arenaB, F(HEAD + 4), F(HEAD + 5), N, 512,
                                                    (float*)d_out);
}
